// AttentionBlock_74483322847742
// MI455X (gfx1250) — compile-verified, matrix operand order checked by reading, not run on hardware
//
#include <hip/hip_runtime.h>
#include <hip/hip_bf16.h>

typedef __attribute__((ext_vector_type(16))) _Float16 v16h;
typedef __attribute__((ext_vector_type(8)))  float    v8f;

#define BATCH 4
#define CHN   256
#define SDIM  4096          // H*W
#define HEADS 4
#define DH    64
#define GROUPS 8
#define CPG   (CHN / GROUPS)   // 32
#define EPSV  1e-5f
#define FRAG  512              // halves per packed 16x32 / 32x16 fragment tile

// ---------------------------------------------------------------- WMMA helpers

__device__ __forceinline__ v8f wmma_f16(v16h a, v16h b, v8f c) {
  return __builtin_amdgcn_wmma_f32_16x16x32_f16(
      /*neg_a=*/false, a, /*neg_b=*/false, b,
      /*c_mod=*/(short)0, c, /*reuse_a=*/false, /*reuse_b=*/false);
}

// Offset (in halves) of matrix element inside a packed fragment tile.
// kk = K index within the 32-wide K slab; rc = row (A-frag) or col (B-frag).
// ISA 7.12.2: lane = ((kk>>3)&1)*16 + rc ; elem = (kk>>4)*8 + (kk&7)
__device__ __forceinline__ int frag_off(int kk, int rc) {
  return (((((kk >> 3) & 1) << 4) | rc) << 4) | (((kk >> 4) << 3) | (kk & 7));
}

// Per-lane contiguous 32B fragment load (2 x b128).
__device__ __forceinline__ v16h load_frag(const _Float16* tile) {
  return *(const v16h*)(tile + ((threadIdx.x & 31) << 4));
}

// ---------------------------------------------------------------- GroupNorm
// Writes xn pre-packed as B-fragments for the QKV GEMM.

__global__ void gn_kernel(const float* __restrict__ x,
                          const float* __restrict__ gamma,
                          const float* __restrict__ beta,
                          _Float16* __restrict__ xnpk) {
  const int bg = blockIdx.x;              // b * GROUPS + g
  const int b  = bg / GROUPS, g = bg % GROUPS;
  const float* xp = x + (size_t)(b * CHN + g * CPG) * SDIM;
  const int n = CPG * SDIM;               // 131072

  float sum = 0.f, sq = 0.f;
  for (int i = threadIdx.x; i < n; i += blockDim.x) {
    float v = xp[i]; sum += v; sq += v * v;
  }
  __shared__ float ssum[256], ssq[256];
  ssum[threadIdx.x] = sum; ssq[threadIdx.x] = sq;
  __syncthreads();
  for (int s = 128; s > 0; s >>= 1) {
    if (threadIdx.x < s) {
      ssum[threadIdx.x] += ssum[threadIdx.x + s];
      ssq[threadIdx.x]  += ssq[threadIdx.x + s];
    }
    __syncthreads();
  }
  const float mean = ssum[0] / (float)n;
  const float var  = ssq[0] / (float)n - mean * mean;
  const float rstd = __frsqrt_rn(var + EPSV);

  for (int i = threadIdx.x; i < n; i += blockDim.x) {
    const int cl = i >> 12;               // i / SDIM
    const int c  = g * CPG + cl;
    const int s  = i & (SDIM - 1);
    const float v = (xp[i] - mean) * rstd;
    xnpk[(((size_t)b * 256 + (s >> 4)) * 8 + (c >> 5)) * FRAG +
         frag_off(c & 31, s & 15)] = (_Float16)(v * gamma[c] + beta[c]);
  }
}

// ---------------------------------------------------------------- weights -> f16 A-fragments

__global__ void cvt_kernel(const float* __restrict__ wq,
                           const float* __restrict__ wp,
                           _Float16* __restrict__ wqpk,
                           _Float16* __restrict__ wppk) {
  const int i = blockIdx.x * blockDim.x + threadIdx.x;
  if (i < 3 * CHN * CHN) {
    const int o = i >> 8, c = i & 255;
    wqpk[(((o >> 4) * 8) + (c >> 5)) * FRAG + frag_off(c & 31, o & 15)] =
        (_Float16)wq[i];
  }
  if (i < CHN * CHN) {
    const int o = i >> 8, c = i & 255;
    wppk[(((o >> 4) * 8) + (c >> 5)) * FRAG + frag_off(c & 31, o & 15)] =
        (_Float16)wp[i];
  }
}

// ---------------------------------------------------------------- QKV GEMM

__global__ void qkv_kernel(const _Float16* __restrict__ Wq,   // packed A tiles
                           const _Float16* __restrict__ xnpk, // packed B tiles
                           const float* __restrict__ bias,
                           _Float16* __restrict__ Qpk,
                           _Float16* __restrict__ Kpk,
                           _Float16* __restrict__ Vpk) {
  const int b    = blockIdx.z;
  const int wave = threadIdx.x >> 5;
  const int lane = threadIdx.x & 31;
  const int m0 = blockIdx.y * 128 + (wave >> 1) * 32;   // output-channel tile
  const int n0 = blockIdx.x * 64  + (wave & 1) * 32;    // pixel tile
  const int mt0 = m0 >> 4, nt0 = n0 >> 4;
  const _Float16* xb = xnpk + (size_t)b * 256 * 8 * FRAG;

  v8f acc[2][2] = {};
#pragma unroll 2
  for (int kt = 0; kt < 8; ++kt) {
    v16h a0 = load_frag(Wq + ((size_t)mt0 * 8 + kt) * FRAG);
    v16h a1 = load_frag(Wq + ((size_t)(mt0 + 1) * 8 + kt) * FRAG);
    v16h b0 = load_frag(xb + ((size_t)nt0 * 8 + kt) * FRAG);
    v16h b1 = load_frag(xb + ((size_t)(nt0 + 1) * 8 + kt) * FRAG);
    acc[0][0] = wmma_f16(a0, b0, acc[0][0]);
    acc[0][1] = wmma_f16(a0, b1, acc[0][1]);
    acc[1][0] = wmma_f16(a1, b0, acc[1][0]);
    acc[1][1] = wmma_f16(a1, b1, acc[1][1]);
  }

  const int colb = lane & 15, rowb = (lane >> 4) << 3;
#pragma unroll
  for (int tm = 0; tm < 2; ++tm)
#pragma unroll
    for (int tn = 0; tn < 2; ++tn)
#pragma unroll
      for (int r = 0; r < 8; ++r) {
        const int o = m0 + tm * 16 + rowb + r;
        const int s = n0 + tn * 16 + colb;
        const _Float16 hv = (_Float16)(acc[tm][tn][r] + bias[o]);
        if (o < CHN) {                                  // Q (A-frag layout)
          const int bh = b * HEADS + (o >> 6), dd = o & 63;
          Qpk[(((size_t)bh * 256 + (s >> 4)) * 2 + (dd >> 5)) * FRAG +
              frag_off(dd & 31, s & 15)] = hv;
        } else if (o < 2 * CHN) {                       // K (B-frag layout)
          const int oo = o - CHN;
          const int bh = b * HEADS + (oo >> 6), dd = oo & 63;
          Kpk[(((size_t)bh * 256 + (s >> 4)) * 2 + (dd >> 5)) * FRAG +
              frag_off(dd & 31, s & 15)] = hv;
        } else {                                        // V (B-frag layout)
          const int oo = o - 2 * CHN;
          const int bh = b * HEADS + (oo >> 6), dd = oo & 63;
          Vpk[(((size_t)bh * 128 + (s >> 5)) * 4 + (dd >> 4)) * FRAG +
              frag_off(s & 31, dd & 15)] = hv;
        }
      }
}

// ---------------------------------------------------------------- Flash attention
// One wave = 32 query rows (2x16 tiles); block = 4 waves = 128 rows.
// Every K/V fragment load now feeds TWO WMMAs (one per query sub-tile).

__global__ void attn_kernel(const _Float16* __restrict__ Qpk,
                            const _Float16* __restrict__ Kpk,
                            const _Float16* __restrict__ Vpk,
                            _Float16* __restrict__ Opk) {
  const int bh   = blockIdx.y;
  const int wave = threadIdx.x >> 5;
  const int lane = threadIdx.x & 31;
  const int q0   = blockIdx.x * 128 + wave * 32;

  const _Float16* Kb = Kpk + (size_t)bh * 256 * 2 * FRAG;
  const _Float16* Vb = Vpk + (size_t)bh * 128 * 4 * FRAG;

  v16h aq[2][2];
#pragma unroll
  for (int qt = 0; qt < 2; ++qt) {
    const _Float16* Qt = Qpk + ((size_t)bh * 256 + (q0 >> 4) + qt) * 2 * FRAG;
    aq[qt][0] = load_frag(Qt);
    aq[qt][1] = load_frag(Qt + FRAG);
  }

  v8f acc[2][4] = {};                        // 2 x (16 x 64) accumulators
  float m_run[2][8], l_run[2][8];
#pragma unroll
  for (int qt = 0; qt < 2; ++qt)
#pragma unroll
    for (int r = 0; r < 8; ++r) { m_run[qt][r] = -1e30f; l_run[qt][r] = 0.f; }

  __shared__ _Float16 pbuf[4][4 * FRAG];     // per-wave packed-P staging (16 KB)
  _Float16* pw = pbuf[wave];

  const float ls = 0.125f * 1.44269504089f;  // d^-0.5 * log2(e)
  const int colb = lane & 15, halfb = (lane >> 4) << 3;

  for (int t0 = 0; t0 < SDIM; t0 += 64) {
    if (t0 + 64 < SDIM)
      __builtin_prefetch(Kb + ((size_t)((t0 >> 4) + 4) * 2) * FRAG, 0, 0);

    // ---- scores: each K fragment pair feeds both query sub-tiles
    const int ntb = t0 >> 4;
    v8f sfr[2][4];
#pragma unroll
    for (int j = 0; j < 4; ++j) {
      v16h b0 = load_frag(Kb + ((size_t)(ntb + j) * 2 + 0) * FRAG);
      v16h b1 = load_frag(Kb + ((size_t)(ntb + j) * 2 + 1) * FRAG);
#pragma unroll
      for (int qt = 0; qt < 2; ++qt) {
        v8f c = {};
        c = wmma_f16(aq[qt][0], b0, c);
        c = wmma_f16(aq[qt][1], b1, c);
        sfr[qt][j] = c;
      }
    }

    // ---- online softmax per query sub-tile
#pragma unroll
    for (int qt = 0; qt < 2; ++qt) {
      float alpha[8];
#pragma unroll
      for (int r = 0; r < 8; ++r) {
        float mx = fmaxf(fmaxf(sfr[qt][0][r], sfr[qt][1][r]),
                         fmaxf(sfr[qt][2][r], sfr[qt][3][r]));
#pragma unroll
        for (int off = 1; off < 16; off <<= 1)
          mx = fmaxf(mx, __shfl_xor(mx, off, 32));
        const float mnew = fmaxf(m_run[qt][r], mx);
        alpha[r] = __builtin_exp2f(ls * (m_run[qt][r] - mnew));
        m_run[qt][r] = mnew;
        float psum = 0.f;
#pragma unroll
        for (int j = 0; j < 4; ++j) {
          const float p = __builtin_exp2f(ls * (sfr[qt][j][r] - mnew));
          sfr[qt][j][r] = p;
          psum += p;
        }
#pragma unroll
        for (int off = 1; off < 16; off <<= 1)
          psum += __shfl_xor(psum, off, 32);
        l_run[qt][r] = l_run[qt][r] * alpha[r] + psum;
      }
#pragma unroll
      for (int n = 0; n < 4; ++n)
#pragma unroll
        for (int r = 0; r < 8; ++r) acc[qt][n][r] *= alpha[r];

      // P: C-layout -> packed A-frag layout in LDS (2 tiles per sub-tile)
#pragma unroll
      for (int j = 0; j < 4; ++j) {
        const int kkey = j * 16 + colb;            // key offset 0..63
        const int kt = kkey >> 5, kk = kkey & 31;
#pragma unroll
        for (int r = 0; r < 8; ++r)
          pw[(qt * 2 + kt) * FRAG + frag_off(kk, halfb + r)] =
              (_Float16)sfr[qt][j][r];
      }
    }

    v16h pa[2][2];
#pragma unroll
    for (int qt = 0; qt < 2; ++qt) {
      pa[qt][0] = load_frag(pw + (qt * 2 + 0) * FRAG);
      pa[qt][1] = load_frag(pw + (qt * 2 + 1) * FRAG);
    }

    // ---- acc += P(32x64) x V(64x64): each V fragment pair feeds both sub-tiles
    const int ktt = t0 >> 5;
#pragma unroll
    for (int n = 0; n < 4; ++n) {
      v16h bv0 = load_frag(Vb + ((size_t)ktt * 4 + n) * FRAG);
      v16h bv1 = load_frag(Vb + ((size_t)(ktt + 1) * 4 + n) * FRAG);
#pragma unroll
      for (int qt = 0; qt < 2; ++qt) {
        acc[qt][n] = wmma_f16(pa[qt][0], bv0, acc[qt][n]);
        acc[qt][n] = wmma_f16(pa[qt][1], bv1, acc[qt][n]);
      }
    }
  }

  // ---- normalize; store pre-packed as proj-GEMM B-fragments
  const int b = bh >> 2, h = bh & 3;
#pragma unroll
  for (int qt = 0; qt < 2; ++qt)
#pragma unroll
    for (int r = 0; r < 8; ++r) {
      const float inv = 1.f / l_run[qt][r];
      const int srow = q0 + qt * 16 + halfb + r;
#pragma unroll
      for (int n = 0; n < 4; ++n) {
        const int c = h * DH + n * 16 + colb;      // channel 0..255
        Opk[(((size_t)b * 256 + (srow >> 4)) * 8 + (c >> 5)) * FRAG +
            frag_off(c & 31, srow & 15)] = (_Float16)(acc[qt][n][r] * inv);
      }
    }
}

// ---------------------------------------------------------------- Proj + residual

__global__ void proj_kernel(const _Float16* __restrict__ Wp,   // packed A tiles
                            const _Float16* __restrict__ Opk,  // packed B tiles
                            const float* __restrict__ bias,
                            const float* __restrict__ x,
                            float* __restrict__ out) {
  const int b    = blockIdx.z;
  const int wave = threadIdx.x >> 5;
  const int lane = threadIdx.x & 31;
  const int m0 = blockIdx.y * 128 + (wave >> 1) * 32;
  const int n0 = blockIdx.x * 64  + (wave & 1) * 32;
  const int mt0 = m0 >> 4, nt0 = n0 >> 4;
  const _Float16* ab = Opk + (size_t)b * 256 * 8 * FRAG;

  v8f acc[2][2] = {};
#pragma unroll 2
  for (int kt = 0; kt < 8; ++kt) {
    v16h a0 = load_frag(Wp + ((size_t)mt0 * 8 + kt) * FRAG);
    v16h a1 = load_frag(Wp + ((size_t)(mt0 + 1) * 8 + kt) * FRAG);
    v16h b0 = load_frag(ab + ((size_t)nt0 * 8 + kt) * FRAG);
    v16h b1 = load_frag(ab + ((size_t)(nt0 + 1) * 8 + kt) * FRAG);
    acc[0][0] = wmma_f16(a0, b0, acc[0][0]);
    acc[0][1] = wmma_f16(a0, b1, acc[0][1]);
    acc[1][0] = wmma_f16(a1, b0, acc[1][0]);
    acc[1][1] = wmma_f16(a1, b1, acc[1][1]);
  }

  const int colb = lane & 15, rowb = (lane >> 4) << 3;
#pragma unroll
  for (int tm = 0; tm < 2; ++tm)
#pragma unroll
    for (int tn = 0; tn < 2; ++tn)
#pragma unroll
      for (int r = 0; r < 8; ++r) {
        const int o = m0 + tm * 16 + rowb + r;
        const int s = n0 + tn * 16 + colb;
        const size_t idx = ((size_t)b * CHN + o) * SDIM + s;
        out[idx] = x[idx] + acc[tm][tn][r] + bias[o];
      }
}

// ---------------------------------------------------------------- launch

extern "C" void kernel_launch(void* const* d_in, const int* in_sizes, int n_in,
                              void* d_out, int out_size, void* d_ws, size_t ws_size,
                              hipStream_t stream) {
  (void)in_sizes; (void)n_in; (void)out_size; (void)ws_size;
  const float* x      = (const float*)d_in[0];
  const float* gamma  = (const float*)d_in[1];
  const float* beta   = (const float*)d_in[2];
  const float* w_qkv  = (const float*)d_in[3];
  const float* b_qkv  = (const float*)d_in[4];
  const float* w_proj = (const float*)d_in[5];
  const float* b_proj = (const float*)d_in[6];
  float* out = (float*)d_out;

  // workspace carve-up (all f16, fragment-packed)
  char* w = (char*)d_ws;
  const size_t tsz = (size_t)BATCH * CHN * SDIM * sizeof(_Float16);  // 8 MB
  _Float16* xnpk = (_Float16*)w; w += tsz;
  _Float16* qpk  = (_Float16*)w; w += tsz;
  _Float16* kpk  = (_Float16*)w; w += tsz;
  _Float16* vpk  = (_Float16*)w; w += tsz;
  _Float16* opk  = (_Float16*)w; w += tsz;
  _Float16* wqpk = (_Float16*)w; w += (size_t)3 * CHN * CHN * sizeof(_Float16);
  _Float16* wppk = (_Float16*)w; w += (size_t)CHN * CHN * sizeof(_Float16);

  // 1. GroupNorm -> packed f16 B-fragments
  gn_kernel<<<dim3(BATCH * GROUPS), dim3(256), 0, stream>>>(x, gamma, beta, xnpk);

  // 2. weights -> packed f16 A-fragments
  cvt_kernel<<<dim3((3 * CHN * CHN + 255) / 256), dim3(256), 0, stream>>>(
      w_qkv, w_proj, wqpk, wppk);

  // 3. QKV GEMM (WMMA)
  qkv_kernel<<<dim3(SDIM / 64, (3 * CHN) / 128, BATCH), dim3(256), 0, stream>>>(
      wqpk, xnpk, b_qkv, qpk, kpk, vpk);

  // 4. flash attention (WMMA, 32 queries/wave)
  attn_kernel<<<dim3(SDIM / 128, BATCH * HEADS), dim3(128), 0, stream>>>(
      qpk, kpk, vpk, opk);

  // 5. proj + residual (WMMA)
  proj_kernel<<<dim3(SDIM / 64, CHN / 128, BATCH), dim3(256), 0, stream>>>(
      wppk, opk, b_proj, x, out);
}
